// GatedGraphConv_26585847562966
// MI455X (gfx1250) — matrix operand order repaired and checked by reading, not compiled, not run on hardware
//
#include <hip/hip_runtime.h>
#include <hip/hip_bf16.h>

// ---------------------------------------------------------------------------
// GatedGraphConv on MI455X (gfx1250, wave32):
//   per t: m_e = h @ W[t,e] + b_w[t,e]           (WMMA bf16 GEMM)
//          m_sum[row] += val * m_e[col]           (f32 global atomics, L2-resident)
//          gi = m_sum @ w_ih^T + b_ih             (WMMA bf16 GEMM, 384 cols)
//          gh = h     @ w_hh^T + b_hh             (WMMA bf16 GEMM, 384 cols)
//          h  = GRU gates(gi, gh, h)              (elementwise)
// ---------------------------------------------------------------------------

typedef __bf16 bf16_t;
typedef __bf16 v16bf __attribute__((ext_vector_type(16)));
typedef __bf16 v8bf  __attribute__((ext_vector_type(8)));
typedef float  v8f   __attribute__((ext_vector_type(8)));

#define D_DIM   128
#define ET_DIM  4
#define LDS_STR 136   // padded row stride (elems): 272B rows, 16B aligned, 4-bank rotation

// ---------------------------------------------------------------------------
// WMMA GEMM: C[M x ldc] (cols chunked by blockIdx.y*128) = A[M x 128] * B[128 x ldb] + bias
// A, B in bf16 (row-major), C f32. 256 threads = 8 waves; block = 128 rows x 128 cols.
// A-row index is CLAMPED (not branched) so the K-loop is exec-mask-free; the
// store guard in the epilogue discards the overhang rows.
// ---------------------------------------------------------------------------
__global__ __launch_bounds__(256) void wmma_gemm_bf16(
    const bf16_t* __restrict__ A,
    const bf16_t* __restrict__ B,
    const float*  __restrict__ bias,
    float* __restrict__ C,
    int M, int ldb, int ldc)
{
    __shared__ bf16_t BsT[128 * LDS_STR];   // BsT[n][k] = B[k][colBase+n]  (34816 B)

    const int tid     = threadIdx.x;
    const int wave    = tid >> 5;
    const int lane    = tid & 31;
    const int half    = (lane >> 4) & 1;
    const int r       = lane & 15;
    const int rowBase = blockIdx.x * 128;
    const int colBase = blockIdx.y * 128;

    // Stage B chunk transposed into LDS. 2048 16B chunks / 256 threads.
    #pragma unroll
    for (int i = 0; i < 8; ++i) {
        int chunk = tid + i * 256;
        int krow  = chunk >> 4;          // 0..127 (K)
        int n0    = (chunk & 15) << 3;   // 0..120 (col within chunk)
        union { uint4 u; bf16_t h[8]; } tbuf;
        tbuf.u = *(const uint4*)(B + (size_t)krow * ldb + colBase + n0);
        #pragma unroll
        for (int j = 0; j < 8; ++j)
            BsT[(n0 + j) * LDS_STR + krow] = tbuf.h[j];
    }
    __syncthreads();

    // Both lane-halves read the same logical row M=r (different K segments).
    // Clamp instead of predicating: keeps EXEC all-ones through the K loop.
    const int arow  = rowBase + wave * 16 + r;
    const int arowc = (arow < M) ? arow : (M - 1);
    const bf16_t* Arow = A + (size_t)arowc * D_DIM;

    v8f acc[8];
    #pragma unroll
    for (int jt = 0; jt < 8; ++jt)
        #pragma unroll
        for (int i = 0; i < 8; ++i) acc[jt][i] = 0.0f;

    #pragma unroll
    for (int k = 0; k < 4; ++k) {                    // K steps of 32
        // A fragment: elem j -> K = 32k + 16*half + j
        v8bf a_lo = *(const v8bf*)(Arow + k * 32 + half * 16);
        v8bf a_hi = *(const v8bf*)(Arow + k * 32 + half * 16 + 8);
        v16bf afrag = __builtin_shufflevector(a_lo, a_hi,
                        0,1,2,3,4,5,6,7,8,9,10,11,12,13,14,15);

        #pragma unroll
        for (int jt = 0; jt < 8; ++jt) {             // 8 col tiles of 16
            // B fragment: elem j -> K = 32k + 16*half + j, N = r  (contiguous in BsT)
            const bf16_t* bp = &BsT[(jt * 16 + r) * LDS_STR + k * 32 + half * 16];
            v8bf b_lo = *(const v8bf*)bp;
            v8bf b_hi = *(const v8bf*)(bp + 8);
            v16bf bfrag = __builtin_shufflevector(b_lo, b_hi,
                            0,1,2,3,4,5,6,7,8,9,10,11,12,13,14,15);
            acc[jt] = __builtin_amdgcn_wmma_f32_16x16x32_bf16(
                          false, afrag, false, bfrag,
                          (short)0, acc[jt], false, false);
        }
    }

    // Epilogue: C/D layout elem i -> M = i + 8*half, N = r. Fuse bias.
    #pragma unroll
    for (int jt = 0; jt < 8; ++jt) {
        int   col = colBase + jt * 16 + r;
        float bv  = bias[col];
        #pragma unroll
        for (int i = 0; i < 8; ++i) {
            int row = rowBase + wave * 16 + i + half * 8;
            if (row < M)
                C[(size_t)row * ldc + col] = acc[jt][i] + bv;
        }
    }
}

// ---------------------------------------------------------------------------
// Sparse aggregation: one wave per edge; msum[row] += val * m[col]
// m_sum (51 MB) is L2-resident on MI455X (192 MB L2) -> fast f32 atomics.
// ---------------------------------------------------------------------------
__global__ __launch_bounds__(256) void scatter_edges(
    const float* __restrict__ m,
    const int*   __restrict__ row,
    const int*   __restrict__ col,
    const float* __restrict__ val,
    float* __restrict__ msum, int E)
{
    int gid  = blockIdx.x * blockDim.x + threadIdx.x;
    int e    = gid >> 5;
    int lane = gid & 31;
    if (e >= E) return;
    int   c  = col[e];
    int   rr = row[e];
    float v  = val[e];
    const float4 mv = *(const float4*)(m + (size_t)c * D_DIM + lane * 4);
    float* dst = msum + (size_t)rr * D_DIM + lane * 4;
    __hip_atomic_fetch_add(dst + 0, v * mv.x, __ATOMIC_RELAXED, __HIP_MEMORY_SCOPE_AGENT);
    __hip_atomic_fetch_add(dst + 1, v * mv.y, __ATOMIC_RELAXED, __HIP_MEMORY_SCOPE_AGENT);
    __hip_atomic_fetch_add(dst + 2, v * mv.z, __ATOMIC_RELAXED, __HIP_MEMORY_SCOPE_AGENT);
    __hip_atomic_fetch_add(dst + 3, v * mv.w, __ATOMIC_RELAXED, __HIP_MEMORY_SCOPE_AGENT);
}

// ---------------------------------------------------------------------------
// Elementwise helpers
// ---------------------------------------------------------------------------
__global__ void f32_to_bf16_k(const float* __restrict__ in, bf16_t* __restrict__ out, int n)
{
    int i = blockIdx.x * blockDim.x + threadIdx.x;
    if (i < n) out[i] = (bf16_t)in[i];
}

// in: [K x 128] f32 -> out: [128 x K] bf16  (transpose, for w_ih/w_hh: gi = m @ W^T)
__global__ void transpose_to_bf16_k(const float* __restrict__ in, bf16_t* __restrict__ out, int K)
{
    int idx = blockIdx.x * blockDim.x + threadIdx.x;
    int total = D_DIM * K;
    if (idx >= total) return;
    int d = idx / K;
    int k = idx - d * K;
    out[idx] = (bf16_t)in[(size_t)k * D_DIM + d];
}

__global__ void zero_f32_k(float* __restrict__ p, int n)
{
    int i = blockIdx.x * blockDim.x + threadIdx.x;
    if (i < n) p[i] = 0.0f;
}

// GRU gates: r = sig(ir+hr), z = sig(iz+hz), n = tanh(in + r*hn), h' = (1-z)*n + z*h
__global__ void gru_gates_k(const float* __restrict__ gi, const float* __restrict__ gh,
                            const float* __restrict__ hprev, float* __restrict__ hout, int n)
{
    int idx = blockIdx.x * blockDim.x + threadIdx.x;
    if (idx >= n) return;
    int nrow = idx >> 7;           // D = 128
    int d    = idx & 127;
    const float* gir = gi + (size_t)nrow * 384;
    const float* ghr = gh + (size_t)nrow * 384;
    float ir  = gir[d],       hr = ghr[d];
    float iz  = gir[128 + d], hz = ghr[128 + d];
    float inn = gir[256 + d], hn = ghr[256 + d];
    float rg = 1.0f / (1.0f + __expf(-(ir + hr)));
    float zg = 1.0f / (1.0f + __expf(-(iz + hz)));
    float ng = tanhf(inn + rg * hn);
    hout[idx] = (1.0f - zg) * ng + zg * hprev[idx];
}

// ---------------------------------------------------------------------------
// Host orchestration
// ---------------------------------------------------------------------------
extern "C" void kernel_launch(void* const* d_in, const int* in_sizes, int n_in,
                              void* d_out, int out_size, void* d_ws, size_t ws_size,
                              hipStream_t stream)
{
    const float* x        = (const float*)d_in[0];
    const int*   edge_row = (const int*)  d_in[1];
    const int*   edge_col = (const int*)  d_in[2];
    const float* edge_val = (const float*)d_in[3];
    const float* weight   = (const float*)d_in[4];
    const float* bias_w   = (const float*)d_in[5];
    const float* w_ih     = (const float*)d_in[6];
    const float* w_hh     = (const float*)d_in[7];
    const float* b_ih     = (const float*)d_in[8];
    const float* b_hh     = (const float*)d_in[9];
    float* out = (float*)d_out;

    const int D  = D_DIM;
    const int ET = ET_DIM;
    const int N  = in_sizes[0] / D;
    const int E  = in_sizes[1] / ET;
    const int T  = in_sizes[4] / (ET * D * D);

    // Workspace carve (256B aligned). Total ~513 MB.
    char* wsp = (char*)d_ws;
    auto carve = [&](size_t bytes) {
        char* p = wsp;
        wsp += (bytes + 255) & ~(size_t)255;
        return (void*)p;
    };
    float*  h_ws   = (float*) carve((size_t)N * D * 4);
    float*  mstage = (float*) carve((size_t)N * D * 4);
    float*  msum   = (float*) carve((size_t)N * D * 4);
    float*  gi     = (float*) carve((size_t)N * 3 * D * 4);
    float*  gh     = (float*) carve((size_t)N * 3 * D * 4);
    bf16_t* h_bf   = (bf16_t*)carve((size_t)N * D * 2);
    bf16_t* ms_bf  = (bf16_t*)carve((size_t)N * D * 2);
    bf16_t* w_bf   = (bf16_t*)carve((size_t)T * ET * D * D * 2);
    bf16_t* wihT   = (bf16_t*)carve((size_t)D * 3 * D * 2);
    bf16_t* whhT   = (bf16_t*)carve((size_t)D * 3 * D * 2);
    (void)ws_size; (void)n_in; (void)out_size;

    const int EW = 256;
    // One-time weight conversions (deterministic; re-done every call)
    int wcount = T * ET * D * D;
    f32_to_bf16_k<<<(wcount + EW - 1) / EW, EW, 0, stream>>>(weight, w_bf, wcount);
    int tcount = D * 3 * D;
    transpose_to_bf16_k<<<(tcount + EW - 1) / EW, EW, 0, stream>>>(w_ih, wihT, 3 * D);
    transpose_to_bf16_k<<<(tcount + EW - 1) / EW, EW, 0, stream>>>(w_hh, whhT, 3 * D);

    const int rowBlocks = (N + 127) / 128;
    const int nd = N * D;

    for (int t = 0; t < T; ++t) {
        const float* hsrc = (t == 0) ? x : h_ws;

        f32_to_bf16_k<<<(nd + EW - 1) / EW, EW, 0, stream>>>(hsrc, h_bf, nd);
        zero_f32_k   <<<(nd + EW - 1) / EW, EW, 0, stream>>>(msum, nd);

        for (int e = 0; e < ET; ++e) {
            const bf16_t* Be = w_bf   + ((size_t)t * ET + e) * D * D;
            const float*  be = bias_w + ((size_t)t * ET + e) * D;
            wmma_gemm_bf16<<<dim3(rowBlocks, 1), 256, 0, stream>>>(
                h_bf, Be, be, mstage, N, D, D);
            int sthreads = E * 32;
            scatter_edges<<<(sthreads + EW - 1) / EW, EW, 0, stream>>>(
                mstage,
                edge_row + (size_t)e * E,
                edge_col + (size_t)e * E,
                edge_val + (size_t)e * E,
                msum, E);
        }

        f32_to_bf16_k<<<(nd + EW - 1) / EW, EW, 0, stream>>>(msum, ms_bf, nd);

        wmma_gemm_bf16<<<dim3(rowBlocks, 3), 256, 0, stream>>>(
            ms_bf, wihT, b_ih, gi, N, 3 * D, 3 * D);
        wmma_gemm_bf16<<<dim3(rowBlocks, 3), 256, 0, stream>>>(
            h_bf, whhT, b_hh, gh, N, 3 * D, 3 * D);

        float* hdst = (t == T - 1) ? out : h_ws;
        gru_gates_k<<<(nd + EW - 1) / EW, EW, 0, stream>>>(gi, gh, hsrc, hdst, nd);
    }
}
